// Decoder_83614423319331
// MI455X (gfx1250) — matrix operand order repaired and checked by reading, not compiled, not run on hardware
//
#include <hip/hip_runtime.h>
#include <hip/hip_bf16.h>
#include <stdint.h>

// ---------------------------------------------------------------------------
// GraphCast-style decoder step on MI455X (gfx1250), bf16 WMMA everywhere.
// ~276 GFLOP of small MLPs; x (72MB) fits in 192MB L2 -> compute bound ->
// run all GEMMs through V_WMMA_F32_16X16X32_BF16.
// N-dim processed in two NT=8 passes (64 acc VGPRs) to stay spill-free.
// ---------------------------------------------------------------------------

typedef __attribute__((ext_vector_type(16))) __bf16    v16bf;
typedef __attribute__((ext_vector_type(8)))  float     v8f;
typedef __attribute__((ext_vector_type(4)))  float     f32x4;
typedef __attribute__((ext_vector_type(4)))  unsigned  u32x4;
typedef __attribute__((ext_vector_type(2)))  unsigned  u32x2;

union Frag { v16bf v; u32x4 q[2]; };

__device__ __forceinline__ unsigned short f2bf(float f) {
  union { float f; unsigned u; } c; c.f = f;
  unsigned u = c.u + (0x7FFFu + ((c.u >> 16) & 1u));   // round-nearest-even
  return (unsigned short)(u >> 16);
}

__device__ __forceinline__ unsigned pk2bf(float lo, float hi) {
  return (unsigned)f2bf(lo) | ((unsigned)f2bf(hi) << 16);
}

// fast SiLU: y * rcp(1 + exp(-y))  (v_exp_f32 + v_rcp_f32, no precise-div chain)
__device__ __forceinline__ float silu(float y) {
  return y * __builtin_amdgcn_rcpf(1.0f + __expf(-y));
}

__device__ __forceinline__ void lds_fence() {
  asm volatile("s_wait_dscnt 0" ::: "memory");
}

template<int NT>
__device__ __forceinline__ void zero_acc(v8f* a) {
  v8f z = {0.f,0.f,0.f,0.f,0.f,0.f,0.f,0.f};
#pragma unroll
  for (int i = 0; i < NT; ++i) a[i] = z;
}

// Accumulate a 16xK (LDS, bf16, row stride ldK) x Kx(NT*16) slice of a
// Kx(NTALL*16) packed-B matrix into acc[NT]. ntoff = first n-tile of slice.
template<int NT, int NTALL, int KSTEPS>
__device__ __forceinline__ void gemm_acc(const unsigned short* tile, int ldK,
                                         const unsigned short* __restrict__ Bp, int ntoff,
                                         v8f* acc, int lane) {
  const int m = lane & 15, kh = lane >> 4;
#pragma unroll
  for (int ks = 0; ks < KSTEPS; ++ks) {
    Frag a;
    const u32x4* ap = (const u32x4*)(tile + m * ldK + ks * 32 + kh * 16);
    a.q[0] = ap[0];            // K = ks*32 + kh*16     .. +7
    a.q[1] = ap[1];            // K = ks*32 + kh*16 + 8 .. +15
#pragma unroll
    for (int nt = 0; nt < NT; ++nt) {
      Frag b;
      const u32x4* bp = (const u32x4*)(Bp + (size_t)(ks * NTALL + ntoff + nt) * 512 + lane * 16);
      b.q[0] = bp[0]; b.q[1] = bp[1];
      acc[nt] = __builtin_amdgcn_wmma_f32_16x16x32_bf16(
          false, a.v, false, b.v, (short)0, acc[nt], false, false);
    }
  }
}

// bias + SiLU, write bf16 tile back to LDS (row stride ldN).
template<int NT>
__device__ __forceinline__ void bias_silu_store(const v8f* acc, const float* __restrict__ bias,
                                                unsigned short* out, int ldN, int lane) {
  const int nl = lane & 15, half = lane >> 4;
#pragma unroll
  for (int nt = 0; nt < NT; ++nt) {
    float b = bias[nt * 16 + nl];
#pragma unroll
    for (int v = 0; v < 8; ++v) {
      float s = silu(acc[nt][v] + b);
      out[(v + 8 * half) * ldN + nt * 16 + nl] = f2bf(s);
    }
  }
}

// ---------------------------------------------------------------------------
// prep kernels
// ---------------------------------------------------------------------------
__global__ void k_zero(float* p, int n) {
  int i = blockIdx.x * blockDim.x + threadIdx.x;
  if (i < n) p[i] = 0.f;
}

__global__ void k_cvt(const float* __restrict__ in, unsigned short* __restrict__ out, int n) {
  int i = blockIdx.x * blockDim.x + threadIdx.x;
  if (i < n) out[i] = f2bf(in[i]);
}

// Pack W[Kreal x Nreal] (row major, stride Nreal) into WMMA B fragments:
// fragment f = (kt * (Npad/16) + nt): 32 lanes x 16 bf16,
// lane l, elem j  ->  W[kt*32 + (l>>4)*16 + j][nt*16 + (l&15)], zero padded.
__global__ void k_pack(const float* __restrict__ W, int Kreal, int Nreal,
                       int Kpad, int Npad, unsigned short* __restrict__ dst) {
  int tid = blockIdx.x * blockDim.x + threadIdx.x;
  int ntiles = Npad >> 4;
  int total = (Kpad >> 5) * ntiles * 32;
  if (tid >= total) return;
  int lane = tid & 31, f = tid >> 5;
  int kt = f / ntiles, nt = f - kt * ntiles;
  int n = nt * 16 + (lane & 15);
  int kb = kt * 32 + (lane >> 4) * 16;
  for (int j = 0; j < 16; ++j) {
    int k = kb + j;
    float v = (k < Kreal && n < Nreal) ? W[(size_t)k * Nreal + n] : 0.f;
    dst[(size_t)f * 512 + lane * 16 + j] = f2bf(v);
  }
}

// ---------------------------------------------------------------------------
// Stage 1: edge encoder  MLP(2 -> 256 -> 256 -> 2) + LN(2)   -> e0 [E,2] f32
// 4 waves / WG, wave owns 16 edge rows.  LDS/wave: tA 1K | tH 8K | tH2 8K | scr 1K
// ---------------------------------------------------------------------------
#define PW_A 18432
__global__ void k_edge_enc(const float* __restrict__ attr,
                           const unsigned short* __restrict__ W1, const float* __restrict__ b1,
                           const unsigned short* __restrict__ W2, const float* __restrict__ b2,
                           const unsigned short* __restrict__ W3, const float* __restrict__ b3,
                           const float* __restrict__ g, const float* __restrict__ be,
                           float* __restrict__ e0, int E) {
  extern __shared__ char smem[];
  const int lane = threadIdx.x & 31, wave = threadIdx.x >> 5;
  const int rowbase = (blockIdx.x * 4 + wave) * 16;
  char* base = smem + wave * PW_A;
  unsigned short* tA  = (unsigned short*)(base);
  unsigned short* tH  = (unsigned short*)(base + 1024);
  unsigned short* tH2 = (unsigned short*)(base + 9216);
  float*          scr = (float*)(base + 17408);

  u32x4 z = {0u,0u,0u,0u};
  ((u32x4*)tA)[lane] = z; ((u32x4*)tA)[lane + 32] = z;          // 16x32 zero
  if (lane < 16) {
    int e = min(rowbase + lane, E - 1);
    tA[lane * 32 + 0] = f2bf(attr[e * 2 + 0]);
    tA[lane * 32 + 1] = f2bf(attr[e * 2 + 1]);
  }
  lds_fence();

#pragma unroll 1
  for (int h = 0; h < 2; ++h) {
    v8f acc[8];
    zero_acc<8>(acc);
    gemm_acc<8, 16, 1>(tA, 32, W1, h * 8, acc, lane);
    bias_silu_store<8>(acc, b1 + h * 128, tH + h * 128, 256, lane);
  }
  lds_fence();

#pragma unroll 1
  for (int h = 0; h < 2; ++h) {
    v8f acc[8];
    zero_acc<8>(acc);
    gemm_acc<8, 16, 8>(tH, 256, W2, h * 8, acc, lane);
    bias_silu_store<8>(acc, b2 + h * 128, tH2 + h * 128, 256, lane);
  }
  lds_fence();

  v8f a3[1]; zero_acc<1>(a3);
  gemm_acc<1, 1, 8>(tH2, 256, W3, 0, a3, lane);
  {
    int nl = lane & 15, half = lane >> 4;
    float b = (nl < 2) ? b3[nl] : 0.f;
#pragma unroll
    for (int v = 0; v < 8; ++v) scr[(v + 8 * half) * 16 + nl] = a3[0][v] + b;
  }
  lds_fence();
  if (lane < 16) {
    int e = rowbase + lane;
    if (e < E) {
      float y0 = scr[lane * 16 + 0], y1 = scr[lane * 16 + 1];
      float mu = 0.5f * (y0 + y1);
      float d0 = y0 - mu, d1 = y1 - mu;
      float rs = rsqrtf(0.5f * (d0 * d0 + d1 * d1) + 1e-5f);
      e0[e * 2 + 0] = d0 * rs * g[0] + be[0];
      e0[e * 2 + 1] = d1 * rs * g[1] + be[1];
    }
  }
}

// ---------------------------------------------------------------------------
// Stage 2: edge processor  MLP(514 -> 256 -> 256 -> 2) + LN(2), residual,
// fused scatter-add into agg via global f32 atomics.
// LDS/wave: tIn 16K | tH 8K | tAe 1K  (tH2 overlays tIn[0:8K], scr overlays tIn[8K:9K])
// ---------------------------------------------------------------------------
#define PW_B 25600
__global__ void k_ep(const unsigned short* __restrict__ xb,
                     const float* __restrict__ e0, const int* __restrict__ eidx,
                     const unsigned short* __restrict__ Ws, const unsigned short* __restrict__ Wd,
                     const unsigned short* __restrict__ We, const float* __restrict__ b1,
                     const unsigned short* __restrict__ W2, const float* __restrict__ b2,
                     const unsigned short* __restrict__ W3, const float* __restrict__ b3,
                     const float* __restrict__ g, const float* __restrict__ be,
                     float* __restrict__ agg, int E) {
  extern __shared__ char smem[];
  const int lane = threadIdx.x & 31, wave = threadIdx.x >> 5;
  const int rowbase = (blockIdx.x * 4 + wave) * 16;
  char* base = smem + wave * PW_B;
  unsigned short* tIn = (unsigned short*)(base);
  unsigned short* tH  = (unsigned short*)(base + 16384);
  unsigned short* tAe = (unsigned short*)(base + 24576);
  unsigned short* tH2 = (unsigned short*)(base);
  float*          scr = (float*)(base + 8192);

  // gather x[src] | x[dst] -> 16 x 512 bf16
  for (int r = 0; r < 16; ++r) {
    int e = min(rowbase + r, E - 1);
    int s = eidx[e], d = eidx[E + e];
    const u32x4* ps = (const u32x4*)(xb + (size_t)s * 256);
    const u32x4* pd = (const u32x4*)(xb + (size_t)d * 256);
    u32x4* pt = (u32x4*)(tIn + r * 512);
    pt[lane]      = ps[lane];
    pt[lane + 32] = pd[lane];
  }
  u32x4 z = {0u,0u,0u,0u};
  ((u32x4*)tAe)[lane] = z; ((u32x4*)tAe)[lane + 32] = z;
  if (lane < 16) {
    int e = min(rowbase + lane, E - 1);
    tAe[lane * 32 + 0] = f2bf(e0[e * 2 + 0]);
    tAe[lane * 32 + 1] = f2bf(e0[e * 2 + 1]);
  }
  lds_fence();

#pragma unroll 1
  for (int h = 0; h < 2; ++h) {
    v8f acc[8];
    zero_acc<8>(acc);
    gemm_acc<8, 16, 8>(tIn,       512, Ws, h * 8, acc, lane);   // x[src] part
    gemm_acc<8, 16, 8>(tIn + 256, 512, Wd, h * 8, acc, lane);   // x[dst] part
    gemm_acc<8, 16, 1>(tAe,        32, We, h * 8, acc, lane);   // edge part (K 2->32)
    bias_silu_store<8>(acc, b1 + h * 128, tH + h * 128, 256, lane);
  }
  lds_fence();

#pragma unroll 1
  for (int h = 0; h < 2; ++h) {
    v8f acc[8];
    zero_acc<8>(acc);
    gemm_acc<8, 16, 8>(tH, 256, W2, h * 8, acc, lane);
    bias_silu_store<8>(acc, b2 + h * 128, tH2 + h * 128, 256, lane);
  }
  lds_fence();

  v8f a3[1]; zero_acc<1>(a3);
  gemm_acc<1, 1, 8>(tH2, 256, W3, 0, a3, lane);
  {
    int nl = lane & 15, half = lane >> 4;
    float b = (nl < 2) ? b3[nl] : 0.f;
#pragma unroll
    for (int v = 0; v < 8; ++v) scr[(v + 8 * half) * 16 + nl] = a3[0][v] + b;
  }
  lds_fence();
  if (lane < 16) {
    int e = rowbase + lane;
    if (e < E) {
      float y0 = scr[lane * 16 + 0], y1 = scr[lane * 16 + 1];
      float mu = 0.5f * (y0 + y1);
      float d0 = y0 - mu, d1 = y1 - mu;
      float rs = rsqrtf(0.5f * (d0 * d0 + d1 * d1) + 1e-5f);
      float ea0 = e0[e * 2 + 0] + d0 * rs * g[0] + be[0];   // residual
      float ea1 = e0[e * 2 + 1] + d1 * rs * g[1] + be[1];
      int d = eidx[E + e];
      atomicAdd(&agg[d * 2 + 0], ea0);
      atomicAdd(&agg[d * 2 + 1], ea1);
    }
  }
}

// ---------------------------------------------------------------------------
// Stage 3: node processor  MLP(258 -> 256 -> 256 -> 256) + LN(256), residual
// -> xnew bf16 [N,256]
// LDS/wave: tX 8K | tH 8K | tAa 1K | tH2 8K   (scr f32 16x256 overlays tX+tH)
// LN epilogue: lane pair (l, l+16) splits a row, f32x4 LDS loads,
// partials combined with __shfl_xor, packed 2xbf16 dword outputs.
// ---------------------------------------------------------------------------
#define PW_C 25600
__global__ void k_np(const unsigned short* __restrict__ xb, const float* __restrict__ agg,
                     const float* __restrict__ xf,
                     const unsigned short* __restrict__ W1x, const unsigned short* __restrict__ W1a,
                     const float* __restrict__ b1,
                     const unsigned short* __restrict__ W2, const float* __restrict__ b2,
                     const unsigned short* __restrict__ W3, const float* __restrict__ b3,
                     const float* __restrict__ g, const float* __restrict__ be,
                     unsigned short* __restrict__ xnew, int Nn) {
  extern __shared__ char smem[];
  const int lane = threadIdx.x & 31, wave = threadIdx.x >> 5;
  const int rowbase = (blockIdx.x * 4 + wave) * 16;
  char* base = smem + wave * PW_C;
  unsigned short* tX  = (unsigned short*)(base);
  unsigned short* tH  = (unsigned short*)(base + 8192);
  unsigned short* tAa = (unsigned short*)(base + 16384);
  unsigned short* tH2 = (unsigned short*)(base + 17408);
  float*          scr = (float*)(base);

  for (int r = 0; r < 16; ++r) {
    int node = min(rowbase + r, Nn - 1);
    ((u32x4*)(tX + r * 256))[lane] = ((const u32x4*)(xb + (size_t)node * 256))[lane];
  }
  u32x4 z = {0u,0u,0u,0u};
  ((u32x4*)tAa)[lane] = z; ((u32x4*)tAa)[lane + 32] = z;
  if (lane < 16) {
    int node = min(rowbase + lane, Nn - 1);
    tAa[lane * 32 + 0] = f2bf(agg[node * 2 + 0]);
    tAa[lane * 32 + 1] = f2bf(agg[node * 2 + 1]);
  }
  lds_fence();

#pragma unroll 1
  for (int h = 0; h < 2; ++h) {
    v8f acc[8];
    zero_acc<8>(acc);
    gemm_acc<8, 16, 8>(tX,  256, W1x, h * 8, acc, lane);
    gemm_acc<8, 16, 1>(tAa,  32, W1a, h * 8, acc, lane);
    bias_silu_store<8>(acc, b1 + h * 128, tH + h * 128, 256, lane);
  }
  lds_fence();

#pragma unroll 1
  for (int h = 0; h < 2; ++h) {
    v8f acc[8];
    zero_acc<8>(acc);
    gemm_acc<8, 16, 8>(tH, 256, W2, h * 8, acc, lane);
    bias_silu_store<8>(acc, b2 + h * 128, tH2 + h * 128, 256, lane);
  }
  lds_fence();

#pragma unroll 1
  for (int h = 0; h < 2; ++h) {
    v8f acc[8];
    zero_acc<8>(acc);
    gemm_acc<8, 16, 8>(tH2, 256, W3, h * 8, acc, lane);
    int nl = lane & 15, half = lane >> 4;
#pragma unroll
    for (int nt = 0; nt < 8; ++nt) {
      float b = b3[(h * 8 + nt) * 16 + nl];
#pragma unroll
      for (int v = 0; v < 8; ++v)
        scr[(v + 8 * half) * 256 + (h * 8 + nt) * 16 + nl] = acc[nt][v] + b;
    }
  }
  lds_fence();
  {
    const int row = lane & 15, half = lane >> 4;
    const int node = rowbase + row;
    if (node < Nn) {
      const f32x4* sv = (const f32x4*)(scr + row * 256 + half * 128);
      float p = 0.f;
#pragma unroll 8
      for (int i = 0; i < 32; ++i) {
        f32x4 v = sv[i];
        p += v.x + v.y + v.z + v.w;
      }
      float mu = (p + __shfl_xor(p, 16, 32)) * (1.f / 256.f);
      float q = 0.f;
#pragma unroll 8
      for (int i = 0; i < 32; ++i) {
        f32x4 v = sv[i];
        f32x4 d = v - mu;
        q += d.x * d.x + d.y * d.y + d.z * d.z + d.w * d.w;
      }
      float var = (q + __shfl_xor(q, 16, 32)) * (1.f / 256.f);
      float rs = rsqrtf(var + 1e-5f);
      const f32x4* gv = (const f32x4*)(g  + half * 128);
      const f32x4* bv = (const f32x4*)(be + half * 128);
      const f32x4* xv = (const f32x4*)(xf + (size_t)node * 256 + half * 128);
      u32x2*       ov = (u32x2*)(xnew + (size_t)node * 256 + half * 128);
#pragma unroll 8
      for (int i = 0; i < 32; ++i) {
        f32x4 v  = sv[i];
        f32x4 gg = gv[i], bb = bv[i], xx = xv[i];
        f32x4 o = (v - mu) * rs * gg + bb + xx;
        u32x2 w;
        w.x = pk2bf(o.x, o.y);
        w.y = pk2bf(o.z, o.w);
        ov[i] = w;
      }
    }
  }
}

// ---------------------------------------------------------------------------
// Stage 4: decoder  MLP(256 -> 128 -> 128 -> 78) + start_features -> out f32
// LDS/wave: tX 8K | tH 4K | tH2 4K
// ---------------------------------------------------------------------------
#define PW_D 16384
__global__ void k_dec(const unsigned short* __restrict__ xnew,
                      const unsigned short* __restrict__ W1, const float* __restrict__ b1,
                      const unsigned short* __restrict__ W2, const float* __restrict__ b2,
                      const unsigned short* __restrict__ W3, const float* __restrict__ b3,
                      const float* __restrict__ start, float* __restrict__ out, int Nn) {
  extern __shared__ char smem[];
  const int lane = threadIdx.x & 31, wave = threadIdx.x >> 5;
  const int rowbase = (blockIdx.x * 4 + wave) * 16;
  char* base = smem + wave * PW_D;
  unsigned short* tX  = (unsigned short*)(base);
  unsigned short* tH  = (unsigned short*)(base + 8192);
  unsigned short* tH2 = (unsigned short*)(base + 12288);

  for (int r = 0; r < 16; ++r) {
    int node = min(rowbase + r, Nn - 1);
    ((u32x4*)(tX + r * 256))[lane] = ((const u32x4*)(xnew + (size_t)node * 256))[lane];
  }
  lds_fence();

  v8f acc[8];
  zero_acc<8>(acc);
  gemm_acc<8, 8, 8>(tX, 256, W1, 0, acc, lane);
  bias_silu_store<8>(acc, b1, tH, 128, lane);
  lds_fence();

  zero_acc<8>(acc);
  gemm_acc<8, 8, 4>(tH, 128, W2, 0, acc, lane);
  bias_silu_store<8>(acc, b2, tH2, 128, lane);
  lds_fence();

  v8f a3[5]; zero_acc<5>(a3);
  gemm_acc<5, 5, 4>(tH2, 128, W3, 0, a3, lane);   // N padded 78 -> 80
  {
    int nl = lane & 15, half = lane >> 4;
#pragma unroll
    for (int nt = 0; nt < 5; ++nt) {
      int n = nt * 16 + nl;
      if (n < 78) {
        float b = b3[n];
#pragma unroll
        for (int v = 0; v < 8; ++v) {
          int node = rowbase + v + 8 * half;
          if (node < Nn)
            out[(size_t)node * 78 + n] = a3[nt][v] + b + start[(size_t)node * 78 + n];
        }
      }
    }
  }
}

// ---------------------------------------------------------------------------
// host: ws layout, weight packing, stage launches
// param-leaf order (jax tree: dicts sorted by key; outer dict insertion order):
//  0 processor_features  1 start_features  2 edge_attr_raw  3 edge_index
//  dec:       4 w(256,128) 5 b  6 w(128,128) 7 b  8 w(128,78) 9 b
//  edge_enc: 10 w(2,256)  11 b 12 w(256,256) 13 b 14 w(256,2) 15 b
//  edge_enc_ln: 16 g 17 beta
//  ep:       18 w(514,256) 19 b 20 w(256,256) 21 b 22 w(256,2) 23 b
//  ep_ln:    24 g 25 beta
//  np:       26 w(258,256) 27 b 28 w(256,256) 29 b 30 w(256,256) 31 b
//  np_ln:    32 g 33 beta
// ---------------------------------------------------------------------------
extern "C" void kernel_launch(void* const* d_in, const int* in_sizes, int n_in,
                              void* d_out, int out_size, void* d_ws, size_t ws_size,
                              hipStream_t stream) {
  (void)n_in; (void)out_size; (void)ws_size;
  const float* xf    = (const float*)d_in[0];
  const float* start = (const float*)d_in[1];
  const float* attr  = (const float*)d_in[2];
  const int*   eidx  = (const int*)d_in[3];
  const float* dec_w1 = (const float*)d_in[4],  *dec_b1 = (const float*)d_in[5];
  const float* dec_w2 = (const float*)d_in[6],  *dec_b2 = (const float*)d_in[7];
  const float* dec_w3 = (const float*)d_in[8],  *dec_b3 = (const float*)d_in[9];
  const float* enc_w1 = (const float*)d_in[10], *enc_b1 = (const float*)d_in[11];
  const float* enc_w2 = (const float*)d_in[12], *enc_b2 = (const float*)d_in[13];
  const float* enc_w3 = (const float*)d_in[14], *enc_b3 = (const float*)d_in[15];
  const float* enc_g  = (const float*)d_in[16], *enc_be = (const float*)d_in[17];
  const float* ep_w1  = (const float*)d_in[18], *ep_b1  = (const float*)d_in[19];
  const float* ep_w2  = (const float*)d_in[20], *ep_b2  = (const float*)d_in[21];
  const float* ep_w3  = (const float*)d_in[22], *ep_b3  = (const float*)d_in[23];
  const float* ep_g   = (const float*)d_in[24], *ep_be  = (const float*)d_in[25];
  const float* np_w1  = (const float*)d_in[26], *np_b1  = (const float*)d_in[27];
  const float* np_w2  = (const float*)d_in[28], *np_b2  = (const float*)d_in[29];
  const float* np_w3  = (const float*)d_in[30], *np_b3  = (const float*)d_in[31];
  const float* np_g   = (const float*)d_in[32], *np_be  = (const float*)d_in[33];

  const int Nn = in_sizes[0] / 256;   // 70682
  const int E  = in_sizes[2] / 2;     // 453600

  char* ws = (char*)d_ws;
  size_t off = 0;
  auto alloc = [&](size_t bytes) -> char* {
    char* p = ws + off;
    off += (bytes + 255) & ~(size_t)255;
    return p;
  };
  unsigned short* xb    = (unsigned short*)alloc((size_t)Nn * 256 * 2);
  unsigned short* xnew  = (unsigned short*)alloc((size_t)Nn * 256 * 2);
  float*          e0    = (float*)alloc((size_t)E * 2 * 4);
  float*          agg   = (float*)alloc((size_t)Nn * 2 * 4);
  unsigned short* encW1p = (unsigned short*)alloc(32  * 256 * 2);
  unsigned short* encW2p = (unsigned short*)alloc(256 * 256 * 2);
  unsigned short* encW3p = (unsigned short*)alloc(256 * 16  * 2);
  unsigned short* epWs   = (unsigned short*)alloc(256 * 256 * 2);
  unsigned short* epWd   = (unsigned short*)alloc(256 * 256 * 2);
  unsigned short* epWe   = (unsigned short*)alloc(32  * 256 * 2);
  unsigned short* epW2p  = (unsigned short*)alloc(256 * 256 * 2);
  unsigned short* epW3p  = (unsigned short*)alloc(256 * 16  * 2);
  unsigned short* npW1x  = (unsigned short*)alloc(256 * 256 * 2);
  unsigned short* npW1a  = (unsigned short*)alloc(32  * 256 * 2);
  unsigned short* npW2p  = (unsigned short*)alloc(256 * 256 * 2);
  unsigned short* npW3p  = (unsigned short*)alloc(256 * 256 * 2);
  unsigned short* decW1p = (unsigned short*)alloc(256 * 128 * 2);
  unsigned short* decW2p = (unsigned short*)alloc(128 * 128 * 2);
  unsigned short* decW3p = (unsigned short*)alloc(128 * 80  * 2);

  // allow >64KB dynamic LDS
  hipFuncSetAttribute((const void*)k_edge_enc, hipFuncAttributeMaxDynamicSharedMemorySize, 4 * PW_A);
  hipFuncSetAttribute((const void*)k_ep,       hipFuncAttributeMaxDynamicSharedMemorySize, 4 * PW_B);
  hipFuncSetAttribute((const void*)k_np,       hipFuncAttributeMaxDynamicSharedMemorySize, 4 * PW_C);
  hipFuncSetAttribute((const void*)k_dec,      hipFuncAttributeMaxDynamicSharedMemorySize, 4 * PW_D);

  // prep
  int n2 = Nn * 2;
  k_zero<<<(n2 + 255) / 256, 256, 0, stream>>>(agg, n2);
  int nx = Nn * 256;
  k_cvt<<<(nx + 255) / 256, 256, 0, stream>>>(xf, xb, nx);

  auto pack = [&](const float* W, int Kr, int Nr, int Kp, int Np, unsigned short* dst) {
    int total = (Kp / 32) * (Np / 16) * 32;
    k_pack<<<(total + 255) / 256, 256, 0, stream>>>(W, Kr, Nr, Kp, Np, dst);
  };
  pack(enc_w1,             2,   256, 32,  256, encW1p);
  pack(enc_w2,             256, 256, 256, 256, encW2p);
  pack(enc_w3,             256, 2,   256, 16,  encW3p);
  pack(ep_w1,              256, 256, 256, 256, epWs);   // rows   0..255 (x[src])
  pack(ep_w1 + 256 * 256,  256, 256, 256, 256, epWd);   // rows 256..511 (x[dst])
  pack(ep_w1 + 512 * 256,  2,   256, 32,  256, epWe);   // rows 512..513 (edge)
  pack(ep_w2,              256, 256, 256, 256, epW2p);
  pack(ep_w3,              256, 2,   256, 16,  epW3p);
  pack(np_w1,              256, 256, 256, 256, npW1x);  // rows   0..255 (x)
  pack(np_w1 + 256 * 256,  2,   256, 32,  256, npW1a);  // rows 256..257 (agg)
  pack(np_w2,              256, 256, 256, 256, npW2p);
  pack(np_w3,              256, 256, 256, 256, npW3p);
  pack(dec_w1,             256, 128, 256, 128, decW1p);
  pack(dec_w2,             128, 128, 128, 128, decW2p);
  pack(dec_w3,             128, 78,  128, 80,  decW3p);

  // stages
  int egw = (E + 63) / 64;
  int ngw = (Nn + 63) / 64;
  k_edge_enc<<<egw, 128, 4 * PW_A, stream>>>(attr, encW1p, enc_b1, encW2p, enc_b2,
                                             encW3p, enc_b3, enc_g, enc_be, e0, E);
  k_ep<<<egw, 128, 4 * PW_B, stream>>>(xb, e0, eidx, epWs, epWd, epWe, ep_b1,
                                       epW2p, ep_b2, epW3p, ep_b3, ep_g, ep_be, agg, E);
  k_np<<<ngw, 128, 4 * PW_C, stream>>>(xb, agg, xf, npW1x, npW1a, np_b1,
                                       npW2p, np_b2, npW3p, np_b3, np_g, np_be, xnew, Nn);
  k_dec<<<ngw, 128, 4 * PW_D, stream>>>(xnew, decW1p, dec_b1, decW2p, dec_b2,
                                        decW3p, dec_b3, start, (float*)d_out, Nn);
}
